// GRU_36593121362031
// MI455X (gfx1250) — matrix operand order repaired and checked by reading, not compiled, not run on hardware
//
#include <hip/hip_runtime.h>
#include <hip/hip_bf16.h>

// Problem constants (match reference)
#define EMB   1024
#define HID   2048
#define BSZ   4
#define SEQ   2048
#define NTOK  (BSZ * SEQ)     // 8192 rows
#define N1    (3 * HID)       // 6144
#define CH    8               // scan chunks per channel
#define CLEN  (SEQ / CH)      // 256 steps per chunk

// GEMM tiling
#define BM 128
#define BN 128
#define BK 64
#define LDST 72               // LDS row stride (halfwords): 64 + 8 pad, rows stay 16B aligned

typedef __attribute__((ext_vector_type(16))) __bf16         v16bf;
typedef __attribute__((ext_vector_type(8)))  float          v8f;
typedef __attribute__((ext_vector_type(4)))  float          f32x4;
typedef __attribute__((ext_vector_type(4)))  unsigned short u16x4;
typedef __attribute__((ext_vector_type(8)))  unsigned short u16x8;

__device__ __forceinline__ unsigned short f32_to_bf16_rne(float f) {
  unsigned u = __builtin_bit_cast(unsigned, f);
  u += 0x7FFFu + ((u >> 16) & 1u);          // round-to-nearest-even
  return (unsigned short)(u >> 16);
}

union Frag { v16bf v; u16x8 h[2]; };

// One-time f32 -> bf16 conversion (vectorized b128 load / b64 store).
__global__ __launch_bounds__(256)
void cvt_bf16(const float* __restrict__ src, unsigned short* __restrict__ dst, int n4) {
  const int i = blockIdx.x * 256 + threadIdx.x;
  if (i < n4) {
    f32x4 f = ((const f32x4*)src)[i];
    u16x4 h;
#pragma unroll
    for (int j = 0; j < 4; ++j) h[j] = f32_to_bf16_rne(f[j]);
    ((u16x4*)dst)[i] = h;
  }
}

// C[M,N] = A[M,K] * B[N,K]^T, all-bf16 operands, f32 accumulate.
// 128x128x64 block tile, 8 waves (2x4), 16 v_wmma per iteration,
// double-buffered LDS -> one workgroup barrier per K-step.
template <int M, int N, int K>
__global__ __launch_bounds__(256)
void wmma_gemm(const unsigned short* __restrict__ Ab,
               const unsigned short* __restrict__ Bb,
               float* __restrict__ Cp) {
  __shared__ __attribute__((aligned(16))) unsigned short As[2][BM * LDST];
  __shared__ __attribute__((aligned(16))) unsigned short Bs[2][BN * LDST];

  const int tid  = threadIdx.x;
  const int lane = tid & 31;
  const int wid  = tid >> 5;
  const int wm   = wid >> 2;                 // 0..1  -> 64 M-rows per wave
  const int wn   = wid & 3;                  // 0..3  -> 32 N-cols per wave
  const int m0   = blockIdx.y * BM;
  const int n0   = blockIdx.x * BN;

  v8f acc[4][2];
#pragma unroll
  for (int mi = 0; mi < 4; ++mi)
#pragma unroll
    for (int ni = 0; ni < 2; ++ni)
#pragma unroll
      for (int r = 0; r < 8; ++r) acc[mi][ni][r] = 0.0f;

  // Per-lane fragment geometry (wave32; ISA 7.12.2 layouts)
  const int lrow = lane & 15;
  const int kA   = (lane < 16) ? 0 : 8;      // A: K {kA..kA+7} and {kA+16..kA+23}
  const int kB   = (lane < 16) ? 0 : 8;      // B: K {kB..kB+7} and {kB+16..kB+23}

  // Staging map: 128x64 bf16 tile = 1024 16B-chunks; 4 chunks per thread.
  int srow[4], scol[4];
#pragma unroll
  for (int i = 0; i < 4; ++i) {
    const int lin = tid + i * 256;           // chunk id 0..1023
    srow[i] = lin >> 3;                      // 8 chunks per 64-wide row
    scol[i] = (lin & 7) << 3;                // element offset within row
  }

  constexpr int NT = K / BK;
  u16x8 ra[4], rb[4];

  // Prologue: fetch tile 0 and stage into buffer 0.
#pragma unroll
  for (int i = 0; i < 4; ++i) {
    ra[i] = *(const u16x8*)(Ab + (size_t)(m0 + srow[i]) * K + scol[i]);
    rb[i] = *(const u16x8*)(Bb + (size_t)(n0 + srow[i]) * K + scol[i]);
  }
#pragma unroll
  for (int i = 0; i < 4; ++i) {
    *(u16x8*)(&As[0][srow[i] * LDST + scol[i]]) = ra[i];
    *(u16x8*)(&Bs[0][srow[i] * LDST + scol[i]]) = rb[i];
  }
  __syncthreads();

  for (int kt = 0; kt < NT; ++kt) {
    const int cur = kt & 1, nxt = cur ^ 1;

    // Fetch tile kt+1 into registers; prefetch tile kt+2 toward L2/L0.
    if (kt + 1 < NT) {
      const int k1 = (kt + 1) * BK;
#pragma unroll
      for (int i = 0; i < 4; ++i) {
        ra[i] = *(const u16x8*)(Ab + (size_t)(m0 + srow[i]) * K + k1 + scol[i]);
        rb[i] = *(const u16x8*)(Bb + (size_t)(n0 + srow[i]) * K + k1 + scol[i]);
      }
    }
    if (kt + 2 < NT) {
      const int k2 = (kt + 2) * BK;
      __builtin_prefetch(Ab + (size_t)(m0 + srow[0]) * K + k2 + scol[0], 0, 1);
      __builtin_prefetch(Bb + (size_t)(n0 + srow[0]) * K + k2 + scol[0], 0, 1);
    }

    // Compute on buffer `cur`: two 32-deep sub-steps, 16 WMMAs.
#pragma unroll
    for (int ks = 0; ks < BK; ks += 32) {
      Frag bf[2];
#pragma unroll
      for (int ni = 0; ni < 2; ++ni) {
        const int rB = wn * 32 + ni * 16 + lrow;
        bf[ni].h[0] = *(const u16x8*)(&Bs[cur][rB * LDST + ks + kB]);
        bf[ni].h[1] = *(const u16x8*)(&Bs[cur][rB * LDST + ks + kB + 16]);
      }
#pragma unroll
      for (int mi = 0; mi < 4; ++mi) {
        Frag af;
        const int rA = wm * 64 + mi * 16 + lrow;
        af.h[0] = *(const u16x8*)(&As[cur][rA * LDST + ks + kA]);
        af.h[1] = *(const u16x8*)(&As[cur][rA * LDST + ks + kA + 16]);
#pragma unroll
        for (int ni = 0; ni < 2; ++ni) {
          acc[mi][ni] = __builtin_amdgcn_wmma_f32_16x16x32_bf16(
              false, af.v, false, bf[ni].v, (short)0, acc[mi][ni], false, false);
        }
      }
    }

    // Stage tile kt+1 into the other buffer. Safe vs. iteration kt-1's reads
    // of buffer `nxt` (separated by the previous barrier); the single barrier
    // below orders these stores before iteration kt+1's reads.
    if (kt + 1 < NT) {
#pragma unroll
      for (int i = 0; i < 4; ++i) {
        *(u16x8*)(&As[nxt][srow[i] * LDST + scol[i]]) = ra[i];
        *(u16x8*)(&Bs[nxt][srow[i] * LDST + scol[i]]) = rb[i];
      }
      __syncthreads();
    }
  }

  // Epilogue: C/D layout -> lane = N col, VGPR r + 8*(lane>=16) = M row.
  // N is compile-time, so r*N folds into immediate store offsets.
  const int moff = (lane >> 4) << 3;
#pragma unroll
  for (int mi = 0; mi < 4; ++mi)
#pragma unroll
    for (int ni = 0; ni < 2; ++ni) {
      float* cptr = Cp + (size_t)(m0 + wm * 64 + mi * 16 + moff) * N +
                    (n0 + wn * 32 + ni * 16 + lrow);
#pragma unroll
      for (int r = 0; r < 8; ++r) cptr[r * N] = acc[mi][ni][r];
    }
}

// Pass 1: per (batch, chunk, channel) compute chunk a-product P and local h L
// (seeded with 0); conv window re-reads raw z across the chunk boundary.
__global__ __launch_bounds__(256)
void scan_pass1(const float* __restrict__ Y, const float* __restrict__ cw,
                float* __restrict__ P, float* __restrict__ L) {
  const int idx = blockIdx.x * 256 + threadIdx.x;   // 65536 threads
  const int hid = idx & (HID - 1);
  const int bc  = idx >> 11;
  const int c   = bc & (CH - 1);
  const int b   = bc >> 3;
  const float w0 = cw[hid * 4 + 0], w1 = cw[hid * 4 + 1];
  const float w2 = cw[hid * 4 + 2], w3 = cw[hid * 4 + 3];
  const float* base = Y + (size_t)b * SEQ * N1;
  const int s0 = c * CLEN;
  float zm1 = (s0 >= 1) ? base[(size_t)(s0 - 1) * N1 + HID + hid] : 0.0f;
  float zm2 = (s0 >= 2) ? base[(size_t)(s0 - 2) * N1 + HID + hid] : 0.0f;
  float zm3 = (s0 >= 3) ? base[(size_t)(s0 - 3) * N1 + HID + hid] : 0.0f;
  float Pa = 1.0f, La = 0.0f;
  for (int s = s0; s < s0 + CLEN; ++s) {
    const float* row = base + (size_t)s * N1;
    const float zp = row[HID + hid];
    const float hp = row[2 * HID + hid];
    const float zc = w0 * zm3 + w1 * zm2 + w2 * zm1 + w3 * zp;
    zm3 = zm2; zm2 = zm1; zm1 = zp;
    const float sg = 1.0f / (1.0f + __expf(-zc));
    const float a  = 1.0f - sg;
    Pa = Pa * a;
    La = a * La + sg * hp;
  }
  P[idx] = Pa;
  L[idx] = La;
}

// Pass 2: fold chunk carries (h = P*h + L over preceding chunks), re-run the
// chunk seeded with the carry, fuse silu(out1) gate, emit bf16 G for GEMM2.
__global__ __launch_bounds__(256)
void scan_pass2(const float* __restrict__ Y, const float* __restrict__ cw,
                const float* __restrict__ P, const float* __restrict__ L,
                unsigned short* __restrict__ G) {
  const int idx = blockIdx.x * 256 + threadIdx.x;
  const int hid = idx & (HID - 1);
  const int bc  = idx >> 11;
  const int c   = bc & (CH - 1);
  const int b   = bc >> 3;
  const float w0 = cw[hid * 4 + 0], w1 = cw[hid * 4 + 1];
  const float w2 = cw[hid * 4 + 2], w3 = cw[hid * 4 + 3];

  float h = 0.0f;
  for (int j = 0; j < c; ++j) {
    const int pi = ((b * CH + j) << 11) + hid;
    h = P[pi] * h + L[pi];
  }

  const float* base = Y + (size_t)b * SEQ * N1;
  const int s0 = c * CLEN;
  float zm1 = (s0 >= 1) ? base[(size_t)(s0 - 1) * N1 + HID + hid] : 0.0f;
  float zm2 = (s0 >= 2) ? base[(size_t)(s0 - 2) * N1 + HID + hid] : 0.0f;
  float zm3 = (s0 >= 3) ? base[(size_t)(s0 - 3) * N1 + HID + hid] : 0.0f;
  for (int s = s0; s < s0 + CLEN; ++s) {
    const float* row = base + (size_t)s * N1;
    const float o1 = row[hid];
    const float zp = row[HID + hid];
    const float hp = row[2 * HID + hid];
    const float zc = w0 * zm3 + w1 * zm2 + w2 * zm1 + w3 * zp;
    zm3 = zm2; zm2 = zm1; zm1 = zp;
    const float sg = 1.0f / (1.0f + __expf(-zc));
    h = (1.0f - sg) * h + sg * hp;
    const float gate = o1 / (1.0f + __expf(-o1));   // silu(out1)
    G[(size_t)(b * SEQ + s) * HID + hid] = f32_to_bf16_rne(gate * h);
  }
}

extern "C" void kernel_launch(void* const* d_in, const int* in_sizes, int n_in,
                              void* d_out, int out_size, void* d_ws, size_t ws_size,
                              hipStream_t stream) {
  const float* x  = (const float*)d_in[0];   // [4,2048,1024]
  const float* w1 = (const float*)d_in[1];   // [6144,1024]
  const float* w2 = (const float*)d_in[2];   // [1024,2048]
  const float* cw = (const float*)d_in[3];   // [2048,4]

  // Workspace layout (~256.5 MB, fully rewritten every call):
  const size_t XB_ELT  = (size_t)NTOK * EMB;     // 8.4M bf16  (16 MB)
  const size_t W1B_ELT = (size_t)N1 * EMB;       // 6.3M bf16  (12 MB)
  const size_t W2B_ELT = (size_t)EMB * HID;      // 2.1M bf16  ( 4 MB)
  const size_t Y_ELT   = (size_t)NTOK * N1;      // f32       (192 MB)
  const size_t G_ELT   = (size_t)NTOK * HID;     // bf16      ( 32 MB)
  const size_t PL_ELT  = (size_t)BSZ * CH * HID; // 65536 f32 each

  char* ws = (char*)d_ws;
  unsigned short* Xb  = (unsigned short*)ws;                    ws += XB_ELT * 2;
  unsigned short* W1b = (unsigned short*)ws;                    ws += W1B_ELT * 2;
  unsigned short* W2b = (unsigned short*)ws;                    ws += W2B_ELT * 2;
  float*          Y   = (float*)ws;                             ws += Y_ELT * 4;
  unsigned short* G   = (unsigned short*)ws;                    ws += G_ELT * 2;
  float*          P   = (float*)ws;                             ws += PL_ELT * 4;
  float*          L   = (float*)ws;

  // One-time bf16 conversions (operands then live in L2 as bf16).
  cvt_bf16<<<(int)(XB_ELT / 4 + 255) / 256, 256, 0, stream>>>(x, Xb, (int)(XB_ELT / 4));
  cvt_bf16<<<(int)(W1B_ELT / 4 + 255) / 256, 256, 0, stream>>>(w1, W1b, (int)(W1B_ELT / 4));
  cvt_bf16<<<(int)(W2B_ELT / 4 + 255) / 256, 256, 0, stream>>>(w2, W2b, (int)(W2B_ELT / 4));

  // GEMM1: Y[8192,6144] = X[8192,1024] * W1[6144,1024]^T
  dim3 g1(N1 / BN, NTOK / BM);
  wmma_gemm<NTOK, N1, EMB><<<g1, 256, 0, stream>>>(Xb, W1b, Y);

  // Conv + sigmoid + chunked linear scan + SiLU gate -> bf16 G[8192,2048]
  scan_pass1<<<(BSZ * CH * HID) / 256, 256, 0, stream>>>(Y, cw, P, L);
  scan_pass2<<<(BSZ * CH * HID) / 256, 256, 0, stream>>>(Y, cw, P, L, G);

  // GEMM2: out[8192,1024] = G[8192,2048] * W2[1024,2048]^T
  dim3 g2(EMB / BN, NTOK / BM);
  wmma_gemm<NTOK, EMB, HID><<<g2, 256, 0, stream>>>(G, W2b, (float*)d_out);
}
